// ComplexAttnMLP_11089605558960
// MI455X (gfx1250) — compile-verified, matrix operand order checked by reading, not run on hardware
//
#include <hip/hip_runtime.h>

// ---------------------------------------------------------------------------
// ComplexAttnMLP for gfx1250 (MI455X): f16 WMMA GEMMs, f32 accumulate.
// B=8, S=1024, IN=MID=OUT=512.
// Each wave computes a 32x32 (cgemm/scores) or 64x32 (attnv) macro-tile.
// 4 accumulators per 16x16 tile (RR, II, RI, IR); the complex combine happens
// in the epilogue, so the K-loop is pure loads + WMMA (no VALU on fragment
// registers -> no WMMA/VALU hazard NOPs).
// ---------------------------------------------------------------------------

typedef _Float16 half8  __attribute__((ext_vector_type(8)));
typedef _Float16 v16h   __attribute__((ext_vector_type(16)));
typedef float    v8f    __attribute__((ext_vector_type(8)));

#define B_DIM  8
#define S_DIM  1024
#define D_DIM  512
#define M_ALL  (B_DIM * S_DIM)          // 8192 rows total
#define RSQRT512 0.04419417382415922f   // 1/sqrt(512)
#define SLOPE 0.01f

union Frag { half8 h8[2]; v16h v; };

// Load one 16x32 f16 WMMA operand fragment for this lane.
// rowptr = start of this lane's row (+k0). Lane half h covers
// K = 8h..8h+7 (VGPR0-3) and K = 16+8h..16+8h+7 (VGPR4-7).
__device__ __forceinline__ v16h load_frag(const _Float16* __restrict__ rowptr, int h) {
    Frag f;
    f.h8[0] = *(const half8*)(rowptr + h * 8);
    f.h8[1] = *(const half8*)(rowptr + 16 + h * 8);
    return f.v;
}

__device__ __forceinline__ v8f wmma_f16(v16h a, v16h b, v8f c) {
    return __builtin_amdgcn_wmma_f32_16x16x32_f16(false, a, false, b, (short)0, c, false, false);
}

// ---------------------------------------------------------------------------
// f32 -> f16 conversion
// ---------------------------------------------------------------------------
__global__ void cvt_f16_kernel(const float* __restrict__ s, _Float16* __restrict__ d, int n) {
    int i = blockIdx.x * blockDim.x + threadIdx.x;
    if (i < n) d[i] = (_Float16)s[i];
}

// ---------------------------------------------------------------------------
// Complex GEMM, 32x32 macro-tile per wave:
//   O[m,n] = sum_k A[m,k]*W[n,k] + bias[n]   (complex)
// A: (M,K) f16 r/i, W: (N,K) f16 r/i.
// ACT: leaky-relu. TRV: write O[b][n][t] (t = m % 1024). F32: fp32 output.
// ---------------------------------------------------------------------------
template <int K, int ACT, int TRV, int F32>
__global__ void __launch_bounds__(128)
cgemm32_kernel(const _Float16* __restrict__ Ar, const _Float16* __restrict__ Ai,
               const _Float16* __restrict__ Wr, const _Float16* __restrict__ Wi,
               const float* __restrict__ br, const float* __restrict__ bi,
               _Float16* __restrict__ Or, _Float16* __restrict__ Oi,
               float* __restrict__ Fr, float* __restrict__ Fi,
               int M, int N) {
    const int lane = threadIdx.x & 31;
    const int wid  = threadIdx.x >> 5;
    const int tilesN = N >> 5;
    const int tile = blockIdx.x * (blockDim.x >> 5) + wid;
    if (tile >= (M >> 5) * tilesN) return;          // wave-uniform guard
    const int m0 = (tile / tilesN) << 5;
    const int n0 = (tile % tilesN) << 5;
    const int h = lane >> 4;
    const int r = lane & 15;

    const _Float16* arp = Ar + (size_t)(m0 + r) * K;
    const _Float16* aip = Ai + (size_t)(m0 + r) * K;
    const _Float16* wrp = Wr + (size_t)(n0 + r) * K;
    const _Float16* wip = Wi + (size_t)(n0 + r) * K;

    v8f aRR[2][2] = {}, aII[2][2] = {}, aRI[2][2] = {}, aIR[2][2] = {};

    for (int k0 = 0; k0 < K; k0 += 32) {
        v16h aR[2], aI[2], wR[2], wI[2];
#pragma unroll
        for (int mi = 0; mi < 2; ++mi) {
            aR[mi] = load_frag(arp + mi * 16 * K + k0, h);
            aI[mi] = load_frag(aip + mi * 16 * K + k0, h);
        }
#pragma unroll
        for (int ni = 0; ni < 2; ++ni) {
            wR[ni] = load_frag(wrp + ni * 16 * K + k0, h);
            wI[ni] = load_frag(wip + ni * 16 * K + k0, h);
        }
#pragma unroll
        for (int mi = 0; mi < 2; ++mi)
#pragma unroll
            for (int ni = 0; ni < 2; ++ni) {
                aRR[mi][ni] = wmma_f16(aR[mi], wR[ni], aRR[mi][ni]);
                aII[mi][ni] = wmma_f16(aI[mi], wI[ni], aII[mi][ni]);
                aRI[mi][ni] = wmma_f16(aR[mi], wI[ni], aRI[mi][ni]);
                aIR[mi][ni] = wmma_f16(aI[mi], wR[ni], aIR[mi][ni]);
            }
    }

#pragma unroll
    for (int mi = 0; mi < 2; ++mi)
#pragma unroll
        for (int ni = 0; ni < 2; ++ni) {
            const int col = n0 + ni * 16 + r;
            const float bre = br[col];
            const float bie = bi[col];
#pragma unroll
            for (int j = 0; j < 8; ++j) {
                const int m = m0 + mi * 16 + j + 8 * h;
                float vr = aRR[mi][ni][j] - aII[mi][ni][j] + bre;
                float vi = aRI[mi][ni][j] + aIR[mi][ni][j] + bie;
                if (ACT) {
                    vr = vr >= 0.0f ? vr : SLOPE * vr;
                    vi = vi >= 0.0f ? vi : SLOPE * vi;
                }
                if (F32) {
                    Fr[(size_t)m * N + col] = vr;
                    Fi[(size_t)m * N + col] = vi;
                } else if (TRV) {
                    const int b = m >> 10, t = m & 1023;
                    const size_t idx = ((size_t)b * N + col) * (size_t)S_DIM + t;
                    Or[idx] = (_Float16)vr;
                    Oi[idx] = (_Float16)vi;
                } else {
                    Or[(size_t)m * N + col] = (_Float16)vr;
                    Oi[(size_t)m * N + col] = (_Float16)vi;
                }
            }
        }
}

// ---------------------------------------------------------------------------
// Scores: Sabs[gm, t] = |sum_d q[gm,d] * k[b*1024+t, d]| / sqrt(512)
// 32x32 macro-tile per wave, per-batch column index.
// ---------------------------------------------------------------------------
__global__ void __launch_bounds__(128)
scores_abs_kernel(const _Float16* __restrict__ Qr, const _Float16* __restrict__ Qi,
                  const _Float16* __restrict__ Kr, const _Float16* __restrict__ Ki,
                  float* __restrict__ Sabs) {
    constexpr int K = D_DIM;
    const int lane = threadIdx.x & 31;
    const int wid  = threadIdx.x >> 5;
    const int tile = blockIdx.x * (blockDim.x >> 5) + wid;
    if (tile >= (M_ALL >> 5) * (S_DIM >> 5)) return;
    const int m0 = (tile / (S_DIM >> 5)) << 5;      // global row
    const int n0 = (tile % (S_DIM >> 5)) << 5;      // t within batch
    const int b  = m0 >> 10;
    const int h = lane >> 4;
    const int r = lane & 15;

    const _Float16* qrp = Qr + (size_t)(m0 + r) * K;
    const _Float16* qip = Qi + (size_t)(m0 + r) * K;
    const _Float16* krp = Kr + (size_t)((b << 10) + n0 + r) * K;
    const _Float16* kip = Ki + (size_t)((b << 10) + n0 + r) * K;

    v8f aRR[2][2] = {}, aII[2][2] = {}, aRI[2][2] = {}, aIR[2][2] = {};

    for (int k0 = 0; k0 < K; k0 += 32) {
        v16h aR[2], aI[2], wR[2], wI[2];
#pragma unroll
        for (int mi = 0; mi < 2; ++mi) {
            aR[mi] = load_frag(qrp + mi * 16 * K + k0, h);
            aI[mi] = load_frag(qip + mi * 16 * K + k0, h);
        }
#pragma unroll
        for (int ni = 0; ni < 2; ++ni) {
            wR[ni] = load_frag(krp + ni * 16 * K + k0, h);
            wI[ni] = load_frag(kip + ni * 16 * K + k0, h);
        }
#pragma unroll
        for (int mi = 0; mi < 2; ++mi)
#pragma unroll
            for (int ni = 0; ni < 2; ++ni) {
                aRR[mi][ni] = wmma_f16(aR[mi], wR[ni], aRR[mi][ni]);
                aII[mi][ni] = wmma_f16(aI[mi], wI[ni], aII[mi][ni]);
                aRI[mi][ni] = wmma_f16(aR[mi], wI[ni], aRI[mi][ni]);
                aIR[mi][ni] = wmma_f16(aI[mi], wR[ni], aIR[mi][ni]);
            }
    }

#pragma unroll
    for (int mi = 0; mi < 2; ++mi)
#pragma unroll
        for (int ni = 0; ni < 2; ++ni) {
            const int col = n0 + ni * 16 + r;
#pragma unroll
            for (int j = 0; j < 8; ++j) {
                const int m = m0 + mi * 16 + j + 8 * h;
                const float sr = aRR[mi][ni][j] - aII[mi][ni][j];
                const float si = aRI[mi][ni][j] + aIR[mi][ni][j];
                Sabs[(size_t)m * S_DIM + col] = sqrtf(sr * sr + si * si) * RSQRT512;
            }
        }
}

// ---------------------------------------------------------------------------
// Row softmax over 1024 elements; one 256-thread block per row; f16 output.
// ---------------------------------------------------------------------------
__global__ void softmax_kernel(const float* __restrict__ S, _Float16* __restrict__ P) {
    __shared__ float red[256];
    const int tid = threadIdx.x;
    const size_t row = blockIdx.x;
    const float* s = S + row * S_DIM;

    float vals[4];
    float mx = -1e30f;
#pragma unroll
    for (int j = 0; j < 4; ++j) {
        vals[j] = s[tid + j * 256];
        mx = fmaxf(mx, vals[j]);
    }
    red[tid] = mx;
    __syncthreads();
    for (int off = 128; off > 0; off >>= 1) {
        if (tid < off) red[tid] = fmaxf(red[tid], red[tid + off]);
        __syncthreads();
    }
    mx = red[0];
    __syncthreads();

    float sum = 0.0f;
#pragma unroll
    for (int j = 0; j < 4; ++j) {
        vals[j] = __expf(vals[j] - mx);
        sum += vals[j];
    }
    red[tid] = sum;
    __syncthreads();
    for (int off = 128; off > 0; off >>= 1) {
        if (tid < off) red[tid] += red[tid + off];
        __syncthreads();
    }
    const float inv = 1.0f / red[0];

    _Float16* p = P + row * S_DIM;
#pragma unroll
    for (int j = 0; j < 4; ++j) p[tid + j * 256] = (_Float16)(vals[j] * inv);
}

// ---------------------------------------------------------------------------
// weighted[gm, d] = sum_t attn[gm, t] * vT[b][d][t]  (real x complex)
// 64x32 macro-tile per wave (4 M-tiles x 2 N-tiles): 8 fragments (16 loads)
// feed 16 WMMAs per K-step.
// ---------------------------------------------------------------------------
__global__ void __launch_bounds__(128)
attnv_kernel(const _Float16* __restrict__ P,
             const _Float16* __restrict__ VTr, const _Float16* __restrict__ VTi,
             _Float16* __restrict__ Wr_, _Float16* __restrict__ Wi_) {
    constexpr int K = S_DIM;
    const int lane = threadIdx.x & 31;
    const int wid  = threadIdx.x >> 5;
    const int tilesN = D_DIM >> 5;                  // 16
    const int tile = blockIdx.x * (blockDim.x >> 5) + wid;
    if (tile >= (M_ALL >> 6) * tilesN) return;
    const int m0 = (tile / tilesN) << 6;            // global row (64-row tile)
    const int n0 = (tile % tilesN) << 5;            // d within batch (32 cols)
    const int b  = m0 >> 10;
    const int h = lane >> 4;
    const int r = lane & 15;

    const _Float16* ap  = P   + (size_t)(m0 + r) * K;
    const _Float16* brp = VTr + ((size_t)b * D_DIM + n0 + r) * (size_t)K;
    const _Float16* bip = VTi + ((size_t)b * D_DIM + n0 + r) * (size_t)K;

    v8f accR[4][2] = {};
    v8f accI[4][2] = {};

    for (int k0 = 0; k0 < K; k0 += 32) {
        v16h a[4], wR[2], wI[2];
#pragma unroll
        for (int mi = 0; mi < 4; ++mi)
            a[mi] = load_frag(ap + mi * 16 * K + k0, h);
#pragma unroll
        for (int ni = 0; ni < 2; ++ni) {
            wR[ni] = load_frag(brp + ni * 16 * K + k0, h);
            wI[ni] = load_frag(bip + ni * 16 * K + k0, h);
        }
#pragma unroll
        for (int mi = 0; mi < 4; ++mi)
#pragma unroll
            for (int ni = 0; ni < 2; ++ni) {
                accR[mi][ni] = wmma_f16(a[mi], wR[ni], accR[mi][ni]);
                accI[mi][ni] = wmma_f16(a[mi], wI[ni], accI[mi][ni]);
            }
    }

#pragma unroll
    for (int mi = 0; mi < 4; ++mi)
#pragma unroll
        for (int ni = 0; ni < 2; ++ni) {
            const int col = n0 + ni * 16 + r;
#pragma unroll
            for (int j = 0; j < 8; ++j) {
                const int m = m0 + mi * 16 + j + 8 * h;
                Wr_[(size_t)m * D_DIM + col] = (_Float16)accR[mi][ni][j];
                Wi_[(size_t)m * D_DIM + col] = (_Float16)accI[mi][ni][j];
            }
        }
}

// ---------------------------------------------------------------------------
// Host-side orchestration
// ---------------------------------------------------------------------------
extern "C" void kernel_launch(void* const* d_in, const int* in_sizes, int n_in,
                              void* d_out, int out_size, void* d_ws, size_t ws_size,
                              hipStream_t stream) {
    (void)in_sizes; (void)n_in; (void)out_size; (void)ws_size;

    const float* x_r   = (const float*)d_in[0];
    const float* x_i   = (const float*)d_in[1];
    const float* l1_wr = (const float*)d_in[2];
    const float* l1_wi = (const float*)d_in[3];
    const float* l1_br = (const float*)d_in[4];
    const float* l1_bi = (const float*)d_in[5];
    const float* q_wr  = (const float*)d_in[6];
    const float* q_wi  = (const float*)d_in[7];
    const float* q_br  = (const float*)d_in[8];
    const float* q_bi  = (const float*)d_in[9];
    const float* k_wr  = (const float*)d_in[10];
    const float* k_wi  = (const float*)d_in[11];
    const float* k_br  = (const float*)d_in[12];
    const float* k_bi  = (const float*)d_in[13];
    const float* v_wr  = (const float*)d_in[14];
    const float* v_wi  = (const float*)d_in[15];
    const float* v_br  = (const float*)d_in[16];
    const float* v_bi  = (const float*)d_in[17];
    const float* l2_wr = (const float*)d_in[18];
    const float* l2_wi = (const float*)d_in[19];
    const float* l2_br = (const float*)d_in[20];
    const float* l2_bi = (const float*)d_in[21];

    const size_t ME = (size_t)M_ALL * D_DIM;   // 4,194,304 (activation elems)
    const size_t WE = (size_t)D_DIM * D_DIM;   // 262,144   (weight elems)
    const size_t AE = (size_t)M_ALL * S_DIM;   // 8,388,608 (attention elems)

    char* w = (char*)d_ws;
    auto alloc_h = [&](size_t elems) { _Float16* p = (_Float16*)w; w += elems * sizeof(_Float16); return p; };
    auto alloc_f = [&](size_t elems) { float* p = (float*)w; w += elems * sizeof(float); return p; };

    _Float16* xh_r = alloc_h(ME);
    _Float16* xh_i = alloc_h(ME);
    _Float16* wl1r = alloc_h(WE); _Float16* wl1i = alloc_h(WE);
    _Float16* wqr  = alloc_h(WE); _Float16* wqi  = alloc_h(WE);
    _Float16* wkr  = alloc_h(WE); _Float16* wki  = alloc_h(WE);
    _Float16* wvr  = alloc_h(WE); _Float16* wvi  = alloc_h(WE);
    _Float16* wl2r = alloc_h(WE); _Float16* wl2i = alloc_h(WE);
    _Float16* y_r  = alloc_h(ME); _Float16* y_i  = alloc_h(ME);
    _Float16* qh_r = alloc_h(ME); _Float16* qh_i = alloc_h(ME);
    _Float16* kh_r = alloc_h(ME); _Float16* kh_i = alloc_h(ME);
    _Float16* vT_r = alloc_h(ME); _Float16* vT_i = alloc_h(ME);
    _Float16* wt_r = alloc_h(ME); _Float16* wt_i = alloc_h(ME);
    _Float16* attn = alloc_h(AE);
    float*    sabs = alloc_f(AE);

    // --- 1. Cast activations and weights to f16 ---
    auto cvt = [&](const float* src, _Float16* dst, size_t n) {
        cvt_f16_kernel<<<(unsigned)((n + 255) / 256), 256, 0, stream>>>(src, dst, (int)n);
    };
    cvt(x_r, xh_r, ME);   cvt(x_i, xh_i, ME);
    cvt(l1_wr, wl1r, WE); cvt(l1_wi, wl1i, WE);
    cvt(q_wr,  wqr,  WE); cvt(q_wi,  wqi,  WE);
    cvt(k_wr,  wkr,  WE); cvt(k_wi,  wki,  WE);
    cvt(v_wr,  wvr,  WE); cvt(v_wi,  wvi,  WE);
    cvt(l2_wr, wl2r, WE); cvt(l2_wi, wl2i, WE);

    const dim3 blk(128);                                              // 4 waves/block
    const unsigned gemm_blocks  = (M_ALL / 32) * (D_DIM / 32) / 4;    // 1024
    const unsigned score_blocks = (M_ALL / 32) * (S_DIM / 32) / 4;    // 2048
    const unsigned attnv_blocks = (M_ALL / 64) * (D_DIM / 32) / 4;    // 512

    // --- 2. y = lrelu(x @ l1^T + b1) ---
    cgemm32_kernel<D_DIM, 1, 0, 0><<<gemm_blocks, blk, 0, stream>>>(
        xh_r, xh_i, wl1r, wl1i, l1_br, l1_bi,
        y_r, y_i, nullptr, nullptr, M_ALL, D_DIM);

    // --- 3. q, k, v (v stored transposed per batch: vT[b][d][t]) ---
    cgemm32_kernel<D_DIM, 0, 0, 0><<<gemm_blocks, blk, 0, stream>>>(
        y_r, y_i, wqr, wqi, q_br, q_bi,
        qh_r, qh_i, nullptr, nullptr, M_ALL, D_DIM);
    cgemm32_kernel<D_DIM, 0, 0, 0><<<gemm_blocks, blk, 0, stream>>>(
        y_r, y_i, wkr, wki, k_br, k_bi,
        kh_r, kh_i, nullptr, nullptr, M_ALL, D_DIM);
    cgemm32_kernel<D_DIM, 0, 1, 0><<<gemm_blocks, blk, 0, stream>>>(
        y_r, y_i, wvr, wvi, v_br, v_bi,
        vT_r, vT_i, nullptr, nullptr, M_ALL, D_DIM);

    // --- 4. |scores|/sqrt(d), softmax rows, attn f16 ---
    scores_abs_kernel<<<score_blocks, blk, 0, stream>>>(qh_r, qh_i, kh_r, kh_i, sabs);
    softmax_kernel<<<M_ALL, 256, 0, stream>>>(sabs, attn);

    // --- 5. weighted = attn @ v ---
    attnv_kernel<<<attnv_blocks, blk, 0, stream>>>(attn, vT_r, vT_i, wt_r, wt_i);

    // --- 6. out = weighted @ l2^T + b2, fp32 to d_out (real block, imag block) ---
    float* out_r = (float*)d_out;
    float* out_i = out_r + ME;
    cgemm32_kernel<D_DIM, 0, 0, 1><<<gemm_blocks, blk, 0, stream>>>(
        wt_r, wt_i, wl2r, wl2i, l2_br, l2_bi,
        nullptr, nullptr, out_r, out_i, M_ALL, D_DIM);
}
